// CausalSelfAttention_28810640621602
// MI455X (gfx1250) — matrix operand order repaired and checked by reading, not compiled, not run on hardware
//
#include <hip/hip_runtime.h>
#include <hip/hip_bf16.h>

typedef __bf16 bf16;
typedef __attribute__((ext_vector_type(16))) __bf16 v16bf;
typedef __attribute__((ext_vector_type(8)))  __bf16 v8bf;
typedef __attribute__((ext_vector_type(8)))  float  v8f;
typedef int v4i __attribute__((vector_size(16)));   // matches builtin's param type

#define WMMA_BF16(A, B, C) \
  __builtin_amdgcn_wmma_f32_16x16x32_bf16(false, (A), false, (B), (short)0, (C), false, false)

// native f32 -> bf16 (RNE); backend emits v_cvt_*bf16* instead of bit-twiddling
__device__ __forceinline__ bf16 f2bf(float x) { return (bf16)x; }

// ---- async global->LDS copy (CDNA5 GLOBAL_LOAD_ASYNC_TO_LDS_B128), guarded ----
#if defined(__has_builtin)
#  if __has_builtin(__builtin_amdgcn_global_load_async_to_lds_b128)
#    define HAVE_ASYNC_LDS 1
#  endif
#endif

#ifdef HAVE_ASYNC_LDS
__device__ __forceinline__ void async_copy_b128(const void* g, void* l) {
  // generic->as1/as3 via integer round-trip (LDS generic addr[31:0] == LDS offset)
  __attribute__((address_space(1))) v4i* gp =
      (__attribute__((address_space(1))) v4i*)(unsigned long long)g;
  __attribute__((address_space(3))) v4i* lp =
      (__attribute__((address_space(3))) v4i*)(unsigned)(unsigned long long)l;
  __builtin_amdgcn_global_load_async_to_lds_b128(gp, lp, 0, 0);
}
__device__ __forceinline__ void async_wait0() {
#if __has_builtin(__builtin_amdgcn_s_wait_asynccnt)
  __builtin_amdgcn_s_wait_asynccnt(0);
#else
  asm volatile("s_wait_asynccnt 0" ::: "memory");
#endif
}
#endif

// ---------------------------------------------------------------------------
// Kernel 1: qkv = x @ w_attn   (M=4096, K=1024, N=3072) bf16 WMMA, fp32 acc.
// Scatters result to q/k/v workspace in [B(2)][H(16)][T(2048)][D(64)] bf16.
// Block tile 128x128, BK=32, 8 waves (4 Mx2 N), each wave 32x64 = 2x4 WMMA tiles.
// ---------------------------------------------------------------------------
__global__ __launch_bounds__(256) void qkv_gemm(const float* __restrict__ x,
                                                const float* __restrict__ w,
                                                bf16* __restrict__ qp,
                                                bf16* __restrict__ kp,
                                                bf16* __restrict__ vp) {
  __shared__ alignas(32) bf16 As[128][32];    // [m][k]
  __shared__ alignas(32) bf16 BsT[128][32];   // [n][k] (transposed for contiguous B frags)
  const int tid = threadIdx.x;
  const int wave = tid >> 5, lane = tid & 31;
  const int lhalf = lane >> 4, l15 = lane & 15;
  const int bm = blockIdx.y, bn = blockIdx.x;
  const int wm = wave & 3, wn = wave >> 2;

  v8f acc[2][4] = {};

  for (int k0 = 0; k0 < 1024; k0 += 32) {
    // stage A tile: 128x32 fp32 -> bf16 LDS
#pragma unroll
    for (int i = 0; i < 4; ++i) {
      int u = tid + 256 * i;                 // 1024 float4 units
      int row = u >> 3, c4 = (u & 7) * 4;
      const float4 f = *(const float4*)(x + (size_t)(bm * 128 + row) * 1024 + k0 + c4);
      union { bf16 b[4]; unsigned long long q; } pk;
      pk.b[0] = f2bf(f.x); pk.b[1] = f2bf(f.y); pk.b[2] = f2bf(f.z); pk.b[3] = f2bf(f.w);
      *(unsigned long long*)&As[row][c4] = pk.q;
    }
    // stage B tile transposed: w[k0+kr][bn*128+c] -> BsT[c][kr]
#pragma unroll
    for (int i = 0; i < 4; ++i) {
      int u = tid + 256 * i;
      int kr = u >> 5, c4 = (u & 31) * 4;
      const float4 f = *(const float4*)(w + (size_t)(k0 + kr) * 3072 + bn * 128 + c4);
      BsT[c4 + 0][kr] = f2bf(f.x);
      BsT[c4 + 1][kr] = f2bf(f.y);
      BsT[c4 + 2][kr] = f2bf(f.z);
      BsT[c4 + 3][kr] = f2bf(f.w);
    }
    __syncthreads();

    v16bf a[2], b[4];
#pragma unroll
    for (int im = 0; im < 2; ++im) {         // A frag: 16 contiguous k at lhalf*16
      int row = wm * 32 + im * 16 + l15;
      union { v16bf v; v8bf h[2]; } ua;
      ua.h[0] = *(const v8bf*)&As[row][lhalf * 16];
      ua.h[1] = *(const v8bf*)&As[row][lhalf * 16 + 8];
      a[im] = ua.v;
    }
#pragma unroll
    for (int in = 0; in < 4; ++in) {         // B frag: 16 contiguous k at lhalf*16
      int col = wn * 64 + in * 16 + l15;
      b[in] = *(const v16bf*)&BsT[col][lhalf * 16];
    }
#pragma unroll
    for (int im = 0; im < 2; ++im)
#pragma unroll
      for (int in = 0; in < 4; ++in)
        acc[im][in] = WMMA_BF16(a[im], b[in], acc[im][in]);
    __syncthreads();
  }

  // scatter to q/k/v [b][h][t][d] bf16
#pragma unroll
  for (int im = 0; im < 2; ++im)
#pragma unroll
    for (int in = 0; in < 4; ++in) {
      int gn = bn * 128 + wn * 64 + in * 16 + l15;
      int sel = gn >> 10, c = gn & 1023, h = c >> 6, d = c & 63;
      bf16* base = (sel == 0) ? qp : ((sel == 1) ? kp : vp);
#pragma unroll
      for (int r = 0; r < 8; ++r) {
        int gm = bm * 128 + wm * 32 + im * 16 + r + 8 * lhalf;
        int bb = gm >> 11, t = gm & 2047;
        base[(((size_t)(bb * 16 + h)) * 2048 + (size_t)t) * 64 + d] = f2bf(acc[im][in][r]);
      }
    }
}

// ---------------------------------------------------------------------------
// Kernel 2: flash attention. One WG per (b, h, 128-query block). 8 waves,
// each wave owns 16 query rows. Streaming softmax, all matmuls via WMMA.
// ---------------------------------------------------------------------------
__global__ __launch_bounds__(256) void attn(const bf16* __restrict__ qp,
                                            const bf16* __restrict__ kp,
                                            const bf16* __restrict__ vp,
                                            bf16* __restrict__ yp) {
  __shared__ alignas(32) bf16 Kt[128][64];        // [key][d]  (B frags for Q.K^T)
  __shared__ alignas(32) bf16 VtT[64][128];       // [d][key]  (B frags for P.V)
  __shared__ alignas(32) bf16 Plds[8][16][128];   // per-wave P (A-layout staging)

  const int tid = threadIdx.x, wave = tid >> 5, lane = tid & 31;
  const int lhalf = lane >> 4, l15 = lane & 15;
  const int bx = blockIdx.x;
  const int qb = bx & 15, h = (bx >> 4) & 15, b = bx >> 8;
  const size_t bh = ((size_t)b * 16 + h) * 2048 * 64;
  const int qbase = qb * 128 + wave * 16;

  // Q A-fragments (K-chunks d[0,32) and d[32,64)) straight from global
  v16bf qf[2];
  {
    const bf16* qrow = qp + bh + (size_t)(qbase + l15) * 64;
#pragma unroll
    for (int f = 0; f < 2; ++f) {
      union { v16bf v; v8bf hh[2]; } u;
      int dd = f * 32 + lhalf * 16;
      u.hh[0] = *(const v8bf*)(qrow + dd);
      u.hh[1] = *(const v8bf*)(qrow + dd + 8);
      qf[f] = u.v;
    }
  }

  v8f o[4] = {};
  float rm[8], rl[8];
#pragma unroll
  for (int r = 0; r < 8; ++r) { rm[r] = -__builtin_inff(); rl[r] = 0.f; }

  for (int kb = 0; kb <= qb; ++kb) {
    const bf16* kbase = kp + bh + (size_t)kb * 128 * 64;
    const bf16* vbase = vp + bh + (size_t)kb * 128 * 64;
    // stage K via async DMA to LDS; V via VGPRs (needs transpose)
#pragma unroll
    for (int i = 0; i < 4; ++i) {
      int u = tid + 256 * i;                 // 1024 x 8-elem units
      int row = u >> 3, seg = (u & 7) * 8;
#ifdef HAVE_ASYNC_LDS
      async_copy_b128(kbase + (size_t)row * 64 + seg, &Kt[row][seg]);
#else
      *(uint4*)&Kt[row][seg] = *(const uint4*)(kbase + (size_t)row * 64 + seg);
#endif
      union { uint4 q; bf16 e[8]; } vv;
      vv.q = *(const uint4*)(vbase + (size_t)row * 64 + seg);
#pragma unroll
      for (int j = 0; j < 8; ++j) VtT[seg + j][row] = vv.e[j];
    }
#ifdef HAVE_ASYNC_LDS
    async_wait0();
#endif
    __syncthreads();

    // S = Q . K^T : 8 key-subtiles x 2 K-chunks
    v8f s[8];
#pragma unroll
    for (int j = 0; j < 8; ++j) {
      v8f a = {};
#pragma unroll
      for (int kc = 0; kc < 2; ++kc) {
        v16bf bfrag = *(const v16bf*)&Kt[j * 16 + l15][kc * 32 + lhalf * 16];
        a = WMMA_BF16(qf[kc], bfrag, a);
      }
      s[j] = a;
    }

    const bool diag = (kb == qb);
#pragma unroll
    for (int r = 0; r < 8; ++r) {
      const int qrow = qbase + r + 8 * lhalf;
      float mx = -__builtin_inff();
#pragma unroll
      for (int j = 0; j < 8; ++j) {
        float sv = s[j][r] * 0.125f;                 // 1/sqrt(64)
        int key = kb * 128 + j * 16 + l15;
        if (diag && key > qrow) sv = -__builtin_inff();
        s[j][r] = sv;
        mx = fmaxf(mx, sv);
      }
#pragma unroll
      for (int off = 8; off >= 1; off >>= 1)
        mx = fmaxf(mx, __shfl_xor(mx, off, 16));
      float mnew = fmaxf(rm[r], mx);
      float alpha = __expf(rm[r] - mnew);
      rm[r] = mnew;
      float ps = 0.f;
#pragma unroll
      for (int j = 0; j < 8; ++j) {
        float p = __expf(s[j][r] - mnew);
        ps += p;
        Plds[wave][r + 8 * lhalf][j * 16 + l15] = f2bf(p);
      }
#pragma unroll
      for (int off = 8; off >= 1; off >>= 1)
        ps += __shfl_xor(ps, off, 16);
      rl[r] = rl[r] * alpha + ps;
#pragma unroll
      for (int n = 0; n < 4; ++n) o[n][r] *= alpha;
    }

    // O += P . V  (per-wave P through LDS; wave-private, LDS in-order => no barrier)
#pragma unroll
    for (int kc = 0; kc < 4; ++kc) {
      union { v16bf v; v8bf hh[2]; } pa;
      pa.hh[0] = *(const v8bf*)&Plds[wave][l15][kc * 32 + lhalf * 16];
      pa.hh[1] = *(const v8bf*)&Plds[wave][l15][kc * 32 + lhalf * 16 + 8];
#pragma unroll
      for (int n = 0; n < 4; ++n) {
        v16bf vf = *(const v16bf*)&VtT[n * 16 + l15][kc * 32 + lhalf * 16];
        o[n] = WMMA_BF16(pa.v, vf, o[n]);
      }
    }
    __syncthreads();
  }

  // normalize (v_rcp + mul) and write y in [B][T][C] bf16 (ready for proj GEMM)
#pragma unroll
  for (int r = 0; r < 8; ++r) {
    float inv = __builtin_amdgcn_rcpf(rl[r]);
    int t = qbase + r + 8 * lhalf;
#pragma unroll
    for (int n = 0; n < 4; ++n) {
      int c = h * 64 + n * 16 + l15;
      yp[((size_t)(b * 2048) + t) * 1024 + c] = f2bf(o[n][r] * inv);
    }
  }
}

// ---------------------------------------------------------------------------
// Kernel 3: out = y @ w_proj  (M=4096, K=1024, N=1024), fp32 output.
// ---------------------------------------------------------------------------
__global__ __launch_bounds__(256) void proj_gemm(const bf16* __restrict__ yp,
                                                 const float* __restrict__ w,
                                                 float* __restrict__ out) {
  __shared__ alignas(32) bf16 As[128][32];
  __shared__ alignas(32) bf16 BsT[128][32];
  const int tid = threadIdx.x;
  const int wave = tid >> 5, lane = tid & 31;
  const int lhalf = lane >> 4, l15 = lane & 15;
  const int bm = blockIdx.y, bn = blockIdx.x;
  const int wm = wave & 3, wn = wave >> 2;

  v8f acc[2][4] = {};

  for (int k0 = 0; k0 < 1024; k0 += 32) {
    // A already bf16: async DMA straight into LDS (no conversion needed)
#pragma unroll
    for (int i = 0; i < 2; ++i) {
      int u = tid + 256 * i;
      int row = u >> 2, seg = (u & 3) * 8;
#ifdef HAVE_ASYNC_LDS
      async_copy_b128(yp + (size_t)(bm * 128 + row) * 1024 + k0 + seg, &As[row][seg]);
#else
      *(uint4*)&As[row][seg] = *(const uint4*)(yp + (size_t)(bm * 128 + row) * 1024 + k0 + seg);
#endif
    }
    // B fp32 -> bf16, transposed
#pragma unroll
    for (int i = 0; i < 4; ++i) {
      int u = tid + 256 * i;
      int kr = u >> 5, c4 = (u & 31) * 4;
      const float4 f = *(const float4*)(w + (size_t)(k0 + kr) * 1024 + bn * 128 + c4);
      BsT[c4 + 0][kr] = f2bf(f.x);
      BsT[c4 + 1][kr] = f2bf(f.y);
      BsT[c4 + 2][kr] = f2bf(f.z);
      BsT[c4 + 3][kr] = f2bf(f.w);
    }
#ifdef HAVE_ASYNC_LDS
    async_wait0();
#endif
    __syncthreads();

    v16bf a[2], b[4];
#pragma unroll
    for (int im = 0; im < 2; ++im) {
      int row = wm * 32 + im * 16 + l15;
      union { v16bf v; v8bf h[2]; } ua;
      ua.h[0] = *(const v8bf*)&As[row][lhalf * 16];
      ua.h[1] = *(const v8bf*)&As[row][lhalf * 16 + 8];
      a[im] = ua.v;
    }
#pragma unroll
    for (int in = 0; in < 4; ++in) {
      int col = wn * 64 + in * 16 + l15;
      b[in] = *(const v16bf*)&BsT[col][lhalf * 16];
    }
#pragma unroll
    for (int im = 0; im < 2; ++im)
#pragma unroll
      for (int in = 0; in < 4; ++in)
        acc[im][in] = WMMA_BF16(a[im], b[in], acc[im][in]);
    __syncthreads();
  }

#pragma unroll
  for (int im = 0; im < 2; ++im)
#pragma unroll
    for (int in = 0; in < 4; ++in) {
      int gn = bn * 128 + wn * 64 + in * 16 + l15;
#pragma unroll
      for (int r = 0; r < 8; ++r) {
        int gm = bm * 128 + wm * 32 + im * 16 + r + 8 * lhalf;
        out[(size_t)gm * 1024 + gn] = acc[im][in][r];
      }
    }
}

// ---------------------------------------------------------------------------
extern "C" void kernel_launch(void* const* d_in, const int* in_sizes, int n_in,
                              void* d_out, int out_size, void* d_ws, size_t ws_size,
                              hipStream_t stream) {
  const float* x      = (const float*)d_in[0];   // [2,2048,1024]
  const float* w_attn = (const float*)d_in[1];   // [1024,3072]
  const float* w_proj = (const float*)d_in[2];   // [1024,1024]
  float* out = (float*)d_out;                    // [2,2048,1024] fp32

  // workspace: q,k,v [B,H,T,D] bf16 (8 MB each) + y [B,T,C] bf16 (8 MB)
  bf16* qp = (bf16*)d_ws;
  bf16* kp = qp + (size_t)4194304;
  bf16* vp = kp + (size_t)4194304;
  bf16* yp = vp + (size_t)4194304;

  dim3 blk(256);
  qkv_gemm<<<dim3(24, 32), blk, 0, stream>>>(x, w_attn, qp, kp, vp);
  attn<<<dim3(512), blk, 0, stream>>>(qp, kp, vp, yp);
  proj_gemm<<<dim3(8, 32), blk, 0, stream>>>(yp, w_proj, out);
}
